// ParticleNetV3_19825569038773
// MI455X (gfx1250) — matrix operand order repaired and checked by reading, not compiled, not run on hardware
//
#include <hip/hip_runtime.h>
#include <cstdint>

#define DEVI __device__ __forceinline__

typedef __attribute__((ext_vector_type(16))) __bf16 v16bf;
typedef __attribute__((ext_vector_type(8)))  float  v8f;
typedef __attribute__((__vector_size__(16))) int    b128_t;   // 16B block for async copy

constexpr int G = 512, P = 128;
constexpr int FN = 16, FG = 8, HID = 128, NCLS = 3;
constexpr int N = G * P;
constexpr int E = 524288;
constexpr float EPS = 1e-5f;
constexpr int TS = 132;          // f32 tile row stride (128 + 4: conflict-free LDS)
constexpr int TILE = P * TS;

// -------- global-memory access through addrspace(1) (forces global_load/store,
// not flat_load; struct-passed pointers otherwise defeat AS inference) --------

typedef const __attribute__((address_space(1))) float*  gcfp;
typedef       __attribute__((address_space(1))) float*  gfp;
typedef const __attribute__((address_space(1))) v16bf*  gcv16p;

DEVI float gload(const float* p) { return *(gcfp)(uintptr_t)p; }
DEVI void  gstore(float* p, float v) { *(gfp)(uintptr_t)p = v; }
DEVI v16bf gload16(const __bf16* p) { return *(gcv16p)(uintptr_t)p; }

// -------- CDNA5 async global->LDS copy (GLOBAL_LOAD_ASYNC_TO_LDS_B128,
// tracked by ASYNCcnt), guarded so compile never breaks --------------------

#if __has_builtin(__builtin_amdgcn_global_load_async_to_lds_b128)
#define ASYNC_LDS_COPY 1
DEVI void async_b128(const void* g, void* l) {
  b128_t* gp = (b128_t*)(uintptr_t)g;                       // generic ptr to 16B block
  b128_t* lp = (b128_t*)l;
  __attribute__((address_space(3))) b128_t* l3 =
      (__attribute__((address_space(3))) b128_t*)lp;        // LDS addrspacecast
  __builtin_amdgcn_global_load_async_to_lds_b128(gp, l3, 0, 0);
}
DEVI void async_wait0() {
#if __has_builtin(__builtin_amdgcn_s_wait_asynccnt)
  __builtin_amdgcn_s_wait_asynccnt(0);
#else
  asm volatile("s_wait_asynccnt 0x0" ::: "memory");
#endif
}
#endif

// ---------------- WMMA helpers (CDNA5 wave32, bf16 -> f32) ----------------

DEVI v8f wmma_bf16(v16bf a, v16bf b, v8f c) {
  return __builtin_amdgcn_wmma_f32_16x16x32_bf16(false, a, false, b, (short)0, c,
                                                 false, false);
}

// A fragment 16x32 from row-major f32 LDS (row stride ld), unguarded.
// lanes 0-15: K 0..15 (elems 0..15); lanes 16-31: K 16..31.
DEVI v16bf a_frag(const float* S, int ld, int row0, int k0, unsigned lane) {
  int r  = row0 + (int)(lane & 15u);
  int kb = k0 + (int)((lane >> 4) * 16u);
  const float* row = S + r * ld + kb;
  v16bf a;
#pragma unroll
  for (int i = 0; i < 8; ++i) {
    a[i]     = (__bf16)row[i];
    a[i + 8] = (__bf16)row[8 + i];
  }
  return a;
}

// B fragment 32x16 from prepped bf16 transposed weights WT[n][Kpad]
// (zero-padded past true K). Lane reads 16 contiguous K -> one 32B global load.
DEVI v16bf b_frag_T(const __bf16* WT, int Kpad, int k0, int col0, unsigned lane) {
  int c  = col0 + (int)(lane & 15u);
  int kb = k0 + (int)((lane >> 4) * 16u);
  return gload16(WT + (size_t)c * Kpad + kb);
}

// Transposed-B fragment from an LDS f32 tile: B[k][n] = S[n][k]  (for H*H^T).
DEVI v16bf bT_frag_lds(const float* S, int ld, int k0, int col0, unsigned lane) {
  int c  = col0 + (int)(lane & 15u);
  int kb = k0 + (int)((lane >> 4) * 16u);
  const float* row = S + c * ld + kb;
  v16bf b;
#pragma unroll
  for (int i = 0; i < 16; ++i) b[i] = (__bf16)row[i];
  return b;
}

// A fragment for EdgeConv z-rows: z row r -> node p=r>>2, neighbor j=knn[..];
// c<128 -> h[p][c]; else h[j][c-128]-h[p][c-128]. Callers unroll kt so the
// c<128 test folds per k-tile (it is uniform within a tile).
DEVI v16bf a_frag_z(const float* cur, const int* knn, int row0, int k0,
                    unsigned lane) {
  int r = row0 + (int)(lane & 15u);
  int p = r >> 2;
  int j = knn[(p << 2) + (r & 3)];
  int kb = k0 + (int)((lane >> 4) * 16u);
  v16bf a;
#pragma unroll
  for (int i = 0; i < 8; ++i) {
    int c1 = kb + i, c2 = kb + 8 + i;
    float f1 = (c1 < 128) ? cur[p * TS + c1]
                          : (cur[j * TS + (c1 - 128)] - cur[p * TS + (c1 - 128)]);
    float f2 = (c2 < 128) ? cur[p * TS + c2]
                          : (cur[j * TS + (c2 - 128)] - cur[p * TS + (c2 - 128)]);
    a[i]     = (__bf16)f1;
    a[i + 8] = (__bf16)f2;
  }
  return a;
}

DEVI float selu_f(float x) {
  const float sc = 1.0507009873554805f, al = 1.6732632423543772f;
  return x > 0.f ? sc * x : sc * al * (expf(x) - 1.f);
}

// ---------------- parameter bundles ----------------

struct ConvP {
  const __bf16 *W1T, *W2T, *W3T, *WscT;          // prepped bf16, [128][Kpad]
  const float *b1, *g1, *be1, *m1, *v1;
  const float *b2, *g2, *be2, *m2, *v2;
  const float *b3, *g3, *be3, *m3, *v3;
  const float *bsc, *gsc, *besc, *msc, *vsc;
};
struct HeadP {
  const float *g0, *b0, *m0, *v0;
  const float *d1W, *d1b, *d2W, *d2b, *oW, *ob;
};

// 128x128x128 GEMM: LDS tile A x prepped bf16 weights, folded bias/relu/BN.
DEVI void gemm128(const float* A, const __bf16* WT, float* D, const float* bias,
                  const float* scl, const float* shf, bool relu, bool accum,
                  int wid, unsigned lane) {
#pragma unroll 2
  for (int t = wid; t < 64; t += 8) {
    int mt = t >> 3, nt = t & 7;
    v8f acc = {};
#pragma unroll
    for (int kt = 0; kt < 4; ++kt) {
      v16bf a = a_frag(A, TS, mt * 16, kt * 32, lane);
      v16bf b = b_frag_T(WT, 128, kt * 32, nt * 16, lane);
      acc = wmma_bf16(a, b, acc);
    }
    int n  = nt * 16 + (int)(lane & 15u);
    int mh = mt * 16 + (int)((lane >> 4) * 8u);
    float bi = bias[n], s = scl[n], sh = shf[n];
#pragma unroll
    for (int r = 0; r < 8; ++r) {
      float x = acc[r] + bi;
      if (relu) x = fmaxf(x, 0.f);
      float y = x * s + sh;
      if (accum) D[(mh + r) * TS + n] += y;
      else       D[(mh + r) * TS + n]  = y;
    }
  }
}

// One DynEdgeConv block, entirely in LDS. cur -> nxt; pooled += colmean(nxt).
__device__ void edge_conv(const ConvP& cp, const float* cur, float* nxt,
                          float* bA, float* bB, float* sq, int* knn,
                          float* lscl, float* lbias, float* pooled,
                          int tid, int wid, unsigned lane) {
  if (tid < 128) {
    float s;
    s = gload(cp.g1 + tid) * rsqrtf(gload(cp.v1 + tid) + EPS);
    lscl[tid]       = s; lscl[128 + tid] = gload(cp.be1 + tid) - gload(cp.m1 + tid) * s;
    lbias[tid]       = gload(cp.b1 + tid);
    s = gload(cp.g2 + tid) * rsqrtf(gload(cp.v2 + tid) + EPS);
    lscl[256 + tid] = s; lscl[384 + tid] = gload(cp.be2 + tid) - gload(cp.m2 + tid) * s;
    lbias[128 + tid] = gload(cp.b2 + tid);
    s = gload(cp.g3 + tid) * rsqrtf(gload(cp.v3 + tid) + EPS);
    lscl[512 + tid] = s; lscl[640 + tid] = gload(cp.be3 + tid) - gload(cp.m3 + tid) * s;
    lbias[256 + tid] = gload(cp.b3 + tid);
    s = gload(cp.gsc + tid) * rsqrtf(gload(cp.vsc + tid) + EPS);
    lscl[768 + tid] = s; lscl[896 + tid] = gload(cp.besc + tid) - gload(cp.msc + tid) * s;
    lbias[384 + tid] = gload(cp.bsc + tid);
    float acc = 0.f;
    const float* row = cur + tid * TS;
    for (int c = 0; c < 128; ++c) { float h = row[c]; acc += h * h; }
    sq[tid] = acc;
  }
  if (tid == 0) {  // L2-resident weights: warm the path (global_prefetch_b8)
    __builtin_prefetch(cp.W1T, 0, 3); __builtin_prefetch(cp.W2T, 0, 3);
    __builtin_prefetch(cp.W3T, 0, 3); __builtin_prefetch(cp.WscT, 0, 3);
  }
  __syncthreads();

  // Gram matrix via WMMA (A = cur, B = cur^T): d2 = sq_p + sq_q - 2*gram.
#pragma unroll 2
  for (int t = wid; t < 64; t += 8) {
    int mt = t >> 3, nt = t & 7;
    v8f acc = {};
#pragma unroll
    for (int kt = 0; kt < 4; ++kt) {
      v16bf a = a_frag(cur, TS, mt * 16, kt * 32, lane);
      v16bf b = bT_frag_lds(cur, TS, kt * 32, nt * 16, lane);
      acc = wmma_bf16(a, b, acc);
    }
    int n  = nt * 16 + (int)(lane & 15u);
    int mh = mt * 16 + (int)((lane >> 4) * 8u);
#pragma unroll
    for (int r = 0; r < 8; ++r) {
      int m = mh + r;
      float d = sq[m] + sq[n] - 2.f * acc[r];
      if (m == n) d += 1e10f;
      bA[m * TS + n] = d;
    }
  }
  __syncthreads();

  // top-4 nearest neighbors per row (stable on ties like lax.top_k)
  if (tid < 128) {
    float v0 = 3e38f, v1 = 3e38f, v2 = 3e38f, v3 = 3e38f;
    int i0 = 0, i1 = 0, i2 = 0, i3 = 0;
    const float* row = bA + tid * TS;
    for (int q = 0; q < 128; ++q) {
      float d = row[q];
      if (d < v3) {
        if      (d < v0) { v3=v2;i3=i2; v2=v1;i2=i1; v1=v0;i1=i0; v0=d;i0=q; }
        else if (d < v1) { v3=v2;i3=i2; v2=v1;i2=i1; v1=d;i1=q; }
        else if (d < v2) { v3=v2;i3=i2; v2=d;i2=q; }
        else             { v3=d;i3=q; }
      }
    }
    knn[tid * 4 + 0] = i0; knn[tid * 4 + 1] = i1;
    knn[tid * 4 + 2] = i2; knn[tid * 4 + 3] = i3;
  }
  __syncthreads();

  // MLP over 512 z-rows, in 4 chunks of 128 rows (32 nodes x 4 neighbors)
  for (int chunk = 0; chunk < 4; ++chunk) {
#pragma unroll 2
    for (int t = wid; t < 64; t += 8) {        // layer1: 128x256x128
      int mt = t >> 3, nt = t & 7;
      v8f acc = {};
#pragma unroll
      for (int kt = 0; kt < 8; ++kt) {
        v16bf a = a_frag_z(cur, knn, chunk * 128 + mt * 16, kt * 32, lane);
        v16bf b = b_frag_T(cp.W1T, 256, kt * 32, nt * 16, lane);
        acc = wmma_bf16(a, b, acc);
      }
      int n  = nt * 16 + (int)(lane & 15u);
      int mh = mt * 16 + (int)((lane >> 4) * 8u);
      float bi = lbias[n], s = lscl[n], sh = lscl[128 + n];
#pragma unroll
      for (int r = 0; r < 8; ++r) {
        float x = fmaxf(acc[r] + bi, 0.f);
        bA[(mh + r) * TS + n] = x * s + sh;
      }
    }
    __syncthreads();
    gemm128(bA, cp.W2T, bB, lbias + 128, lscl + 256, lscl + 384, true, false, wid, lane);
    __syncthreads();
    gemm128(bB, cp.W3T, bA, lbias + 256, lscl + 512, lscl + 640, true, false, wid, lane);
    __syncthreads();
    for (int o = tid; o < 32 * 128; o += 256) {  // max over 4 neighbor rows
      int nl = o >> 7, c = o & 127;
      const float* rb = bA + (nl * 4) * TS + c;
      float mx = fmaxf(fmaxf(rb[0], rb[TS]), fmaxf(rb[2 * TS], rb[3 * TS]));
      nxt[(chunk * 32 + nl) * TS + c] = mx;
    }
    __syncthreads();
  }

  // shortcut: nxt += bn(cur @ Wsc + bsc)
  gemm128(cur, cp.WscT, nxt, lbias + 384, lscl + 768, lscl + 896, false, true, wid, lane);
  __syncthreads();

  if (tid < 128) {
    float s = 0.f;
    for (int p = 0; p < 128; ++p) s += nxt[p * TS + tid];
    pooled[tid] += s * (1.f / 128.f);
  }
  __syncthreads();
}

// ---------------- kernels ----------------

__global__ void k_init(float* p, long n) {
  long i = (long)blockIdx.x * 256 + threadIdx.x;
  if (i < n) gstore(p + i, 0.f);
}

// Transpose + f32->bf16 weight prep: src[K][128] -> dst[n][Kpad] (zero pad k>=K)
__global__ void k_wprep(const float* src, __bf16* dst, int K, int Kpad, int total) {
  int i = blockIdx.x * 256 + threadIdx.x;
  if (i >= total) return;
  int n = i / Kpad, k = i % Kpad;
  __bf16 v = (k < K) ? (__bf16)gload(src + k * 128 + n) : (__bf16)0.0f;
  *((__attribute__((address_space(1))) __bf16*)(uintptr_t)(dst + (size_t)n * Kpad + k)) = v;
}

// GroupNorm + q/k/v/skip projections (K=16 zero-padded to 32: one WMMA/tile)
__global__ __launch_bounds__(256) void k_gnproj(
    const float* x, const float* gw, const float* gb, const float* gms,
    const __bf16* projT,           // [4][128][32] bf16, zero-padded
    const float* bq, const float* bk, const float* bv, const float* bs,
    float* qout /* q,k,v,xr contiguous at stride N*HID */) {
  constexpr int XS = 36;                       // 32 K + pad, conflict-free
  __shared__ float X[128 * XS];
  __shared__ float red[256];
  __shared__ float mu[16], rstd[16], wsh[16], bsh[16], mssh[16];
  __shared__ float bc[512];
  int tid = threadIdx.x, g = blockIdx.x;
  size_t base = (size_t)g * P;
  if (tid < 16) { wsh[tid] = gload(gw + tid); bsh[tid] = gload(gb + tid); mssh[tid] = gload(gms + tid); }
  if (tid < 128) {
    bc[tid]       = gload(bq + tid); bc[128 + tid] = gload(bk + tid);
    bc[256 + tid] = gload(bv + tid); bc[384 + tid] = gload(bs + tid);
  }
  for (int i = tid; i < 128 * XS; i += 256) X[i] = 0.f;   // zero K-pad region
  __syncthreads();
#ifdef ASYNC_LDS_COPY
  // async global->LDS: 4x b128 per node row (16 floats), 16B-aligned both sides
  for (int i = tid; i < 128 * 4; i += 256) {
    int p = i >> 2, f4 = (i & 3) * 4;
    async_b128(x + (base + p) * FN + f4, &X[p * XS + f4]);
  }
  async_wait0();
#else
  for (int i = tid; i < P * FN; i += 256) {
    int p = i >> 4, f = i & 15;
    X[p * XS + f] = gload(x + (base + p) * FN + f);
  }
#endif
  __syncthreads();
  int f = tid & 15, s = tid >> 4;
  { float a = 0.f;
    for (int r = 0; r < 8; ++r) a += X[(s * 8 + r) * XS + f];
    red[s * 16 + f] = a; }
  __syncthreads();
  if (tid < 16) {
    float a = 0.f;
    for (int s2 = 0; s2 < 16; ++s2) a += red[s2 * 16 + tid];
    mu[tid] = a * (1.f / 128.f);
  }
  __syncthreads();
  { float msmu = mssh[f] * mu[f]; float a = 0.f;
    for (int r = 0; r < 8; ++r) {
      int p = s * 8 + r;
      float xs = X[p * XS + f] - msmu;
      X[p * XS + f] = xs; a += xs * xs;
    }
    red[s * 16 + f] = a; }
  __syncthreads();
  if (tid < 16) {
    float a = 0.f;
    for (int s2 = 0; s2 < 16; ++s2) a += red[s2 * 16 + tid];
    rstd[tid] = rsqrtf(a * (1.f / 128.f) + EPS);
  }
  __syncthreads();
  { float sc = wsh[f] * rstd[f], bb = bsh[f];
    for (int r = 0; r < 8; ++r) { int p = s * 8 + r; X[p * XS + f] = X[p * XS + f] * sc + bb; } }
  __syncthreads();

  int wid = tid >> 5; unsigned lane = tid & 31u;
  for (int t = wid; t < 256; t += 8) {
    int proj = t >> 6, tt = t & 63, mt = tt >> 3, nt = tt & 7;
    v8f acc = {};
    v16bf a = a_frag(X, XS, mt * 16, 0, lane);               // K 16..31 are zeros
    v16bf b = b_frag_T(projT + proj * (128 * 32), 32, 0, nt * 16, lane);
    acc = wmma_bf16(a, b, acc);
    int n  = nt * 16 + (int)(lane & 15u);
    int mh = mt * 16 + (int)((lane >> 4) * 8u);
    float bi = bc[proj * 128 + n];
    float* dst = qout + (size_t)proj * ((size_t)N * HID);
#pragma unroll
    for (int r = 0; r < 8; ++r)
      gstore(dst + (base + mh + r) * HID + n, acc[r] + bi);
  }
}

// wave-per-edge: logit = q[dst].k[src]/sqrt(128); segment-max via key atomicMax
__global__ __launch_bounds__(256) void k_logits(const float* q, const float* k,
                                                const int* src, const int* dst,
                                                float* logit, unsigned* mkey) {
  int tid = threadIdx.x; unsigned lane = tid & 31u; int wid = tid >> 5;
  int e = blockIdx.x * 8 + wid;
  if (e >= E) return;
  int s = src[e], d = dst[e];
  const float* qr = q + (size_t)d * HID;
  const float* kr = k + (size_t)s * HID;
  float acc = 0.f;
#pragma unroll
  for (int j = 0; j < 4; ++j) { int c = (int)lane + 32 * j; acc += gload(qr + c) * gload(kr + c); }
  for (int m = 16; m > 0; m >>= 1) acc += __shfl_xor(acc, m, 32);
  if (lane == 0) {
    float l = acc * 0.088388347648318447f;  // 1/sqrt(128)
    gstore(logit + e, l);
    unsigned u = __float_as_uint(l);
    unsigned key = (u & 0x80000000u) ? ~u : (u | 0x80000000u);
    atomicMax(mkey + d, key);
  }
}

// wave-per-edge: ex = exp(l - m); den += ex; agg += ex * v[src]
__global__ __launch_bounds__(256) void k_attn(const float* logit,
                                              const unsigned* mkey, const float* v,
                                              const int* src, const int* dst,
                                              float* den, float* aggub) {
  int tid = threadIdx.x; unsigned lane = tid & 31u; int wid = tid >> 5;
  int e = blockIdx.x * 8 + wid;
  if (e >= E) return;
  int s = src[e], d = dst[e];
  unsigned key = mkey[d];
  unsigned u = (key & 0x80000000u) ? (key & 0x7fffffffu) : ~key;
  float m = __uint_as_float(u);
  float ex = expf(gload(logit + e) - m);
  if (lane == 0) atomicAdd(den + d, ex);
  const float* vr = v + (size_t)s * HID;
  float* ar = aggub + (size_t)d * HID;
#pragma unroll
  for (int j = 0; j < 4; ++j) { int c = (int)lane + 32 * j; atomicAdd(ar + c, ex * gload(vr + c)); }
}

// Per-graph fused: beta-gate -> conv1 -> 2x DynEdgeConv -> pool -> head
__global__ __launch_bounds__(256) void k_graph(const float* aggub, const float* den,
                                               const float* xr, const float* betaW,
                                               const float* ginput, ConvP c2, ConvP c3,
                                               HeadP hp, float* out) {
  extern __shared__ float sm[];
  float* curBuf = sm;
  float* nxtBuf = sm + TILE;
  float* bA     = sm + 2 * TILE;
  float* bB     = sm + 3 * TILE;
  float* sq     = sm + 4 * TILE;              // 128
  int*   knn    = (int*)(sq + 128);           // 512
  float* lscl   = (float*)(knn + 512);        // 1024
  float* lbias  = lscl + 1024;                // 512
  float* pooled = lbias + 512;                // 160
  float* hscr   = pooled + 160;               // 64

  int tid = threadIdx.x, wid = tid >> 5, g = blockIdx.x;
  unsigned lane = tid & 31u;
  size_t base = (size_t)g * P;

  if (tid < 160) pooled[tid] = 0.f;
#ifdef ASYNC_LDS_COPY
  // xr tile: pure copy -> async global->LDS b128 (ASYNCcnt path), overlapped
  // with the aggn loads below. (TS*4 = 528 bytes/row, 16B aligned.)
  for (int i = tid; i < P * (HID / 4); i += 256) {
    int p = i >> 5, c4 = (i & 31) * 4;
    async_b128(xr + (base + p) * HID + c4, &bB[p * TS + c4]);
  }
#endif
  for (int i = tid; i < P * HID; i += 256) {
    int p = i >> 7, c = i & 127;
    float dd = gload(den + base + p);
    bA[p * TS + c] = (dd > 0.f) ? gload(aggub + (base + p) * HID + c) / dd : 0.f;
#ifndef ASYNC_LDS_COPY
    bB[p * TS + c] = gload(xr + (base + p) * HID + c);
#endif
  }
#ifdef ASYNC_LDS_COPY
  async_wait0();
#endif
  __syncthreads();
  if (tid < 128) {
    float acc = 0.f;
    const float* ar  = bA + tid * TS;
    const float* xrr = bB + tid * TS;
    for (int c = 0; c < 128; ++c) {
      float a = ar[c], x = xrr[c];
      acc += a * gload(betaW + c) + x * gload(betaW + 128 + c) + (a - x) * gload(betaW + 256 + c);
    }
    float bta = 1.f / (1.f + expf(-acc));
    float* cr = curBuf + tid * TS;
    for (int c = 0; c < 128; ++c) {
      float v = bta * xrr[c] + (1.f - bta) * ar[c];
      cr[c] = (v > 0.f) ? v : (expf(v) - 1.f);  // elu
    }
  }
  __syncthreads();
  if (tid < 128) {
    float s = 0.f;
    for (int p = 0; p < 128; ++p) s += curBuf[p * TS + tid];
    pooled[tid] += s * (1.f / 128.f);
  }
  __syncthreads();

  edge_conv(c2, curBuf, nxtBuf, bA, bB, sq, knn, lscl, lbias, pooled, tid, wid, lane);
  edge_conv(c3, nxtBuf, curBuf, bA, bB, sq, knn, lscl, lbias, pooled, tid, wid, lane);

  // head
  if (tid < FG) pooled[128 + tid] = gload(ginput + g * FG + tid);
  __syncthreads();
  if (tid < HID + FG) {
    float s = gload(hp.g0 + tid) * rsqrtf(gload(hp.v0 + tid) + EPS);
    pooled[tid] = (pooled[tid] - gload(hp.m0 + tid)) * s + gload(hp.b0 + tid);
  }
  __syncthreads();
  if (tid < 128) {
    float acc = gload(hp.d1b + tid);
    for (int c = 0; c < HID + FG; ++c) acc += pooled[c] * gload(hp.d1W + c * 128 + tid);
    bA[tid] = selu_f(acc);
  }
  __syncthreads();
  if (tid < 128) {
    float acc = gload(hp.d2b + tid);
    for (int c = 0; c < 128; ++c) acc += bA[c] * gload(hp.d2W + c * 128 + tid);
    bB[tid] = selu_f(acc);
  }
  __syncthreads();
  if (tid < NCLS) {
    float acc = gload(hp.ob + tid);
    for (int c = 0; c < 128; ++c) acc += bB[c] * gload(hp.oW + c * NCLS + tid);
    hscr[tid] = acc;
  }
  __syncthreads();
  if (tid == 0) {
    float m = fmaxf(hscr[0], fmaxf(hscr[1], hscr[2]));
    float s = expf(hscr[0] - m) + expf(hscr[1] - m) + expf(hscr[2] - m);
    hscr[8] = m + logf(s);
  }
  __syncthreads();
  if (tid < NCLS) gstore(out + g * NCLS + tid, hscr[tid] - hscr[8]);
}

// ---------------- host ----------------

extern "C" void kernel_launch(void* const* d_in, const int* in_sizes, int n_in,
                              void* d_out, int out_size, void* d_ws, size_t ws_size,
                              hipStream_t stream) {
  (void)in_sizes; (void)n_in; (void)out_size; (void)ws_size;
  auto F = [&](int i) { return (const float*)d_in[i]; };
  const int* ei = (const int*)d_in[1];
  const int* src = ei;
  const int* dst = ei + E;

  float* ws = (float*)d_ws;
  size_t NF = (size_t)N * HID;
  float* q     = ws;                       // q,k,v,xr contiguous (k_gnproj relies on it)
  float* kk    = ws + NF;
  float* vv    = ws + 2 * NF;
  float* xr    = ws + 3 * NF;
  float* aggub = ws + 4 * NF;
  float* den   = ws + 5 * NF;
  unsigned* mkey = (unsigned*)(den + N);
  float* logit   = (float*)(mkey + N);

  size_t woff = 5 * NF + 2 * (size_t)N + (size_t)E;
  woff = (woff + 31) & ~(size_t)31;        // 128B align for 32B vector loads
  __bf16* projT = (__bf16*)(ws + woff);    // 4 * 128*32
  __bf16* c2w   = projT + 4 * 128 * 32;    // W1T 128*256 | W2T | W3T | WscT
  __bf16* c3w   = c2w + 128 * 256 + 3 * 128 * 128;

  long zcount = (long)NF + 2L * N;         // aggub + den + mkey, contiguous
  k_init<<<(int)((zcount + 255) / 256), 256, 0, stream>>>(aggub, zcount);

  // weight prep: f32 [K][128] -> bf16 [128][Kpad] (zero-padded)
  auto prep = [&](const float* s, __bf16* d, int K, int Kpad) {
    int total = 128 * Kpad;
    k_wprep<<<(total + 255) / 256, 256, 0, stream>>>(s, d, K, Kpad, total);
  };
  prep(F(7),  projT,              16, 32);   // q
  prep(F(9),  projT + 128 * 32,   16, 32);   // k
  prep(F(11), projT + 2*128*32,   16, 32);   // v
  prep(F(13), projT + 3*128*32,   16, 32);   // skip
  __bf16* c2W1 = c2w;                __bf16* c2W2 = c2w + 128*256;
  __bf16* c2W3 = c2W2 + 128*128;     __bf16* c2Ws = c2W3 + 128*128;
  prep(F(16), c2W1, 256, 256); prep(F(22), c2W2, 128, 128);
  prep(F(28), c2W3, 128, 128); prep(F(34), c2Ws, 128, 128);
  __bf16* c3W1 = c3w;                __bf16* c3W2 = c3w + 128*256;
  __bf16* c3W3 = c3W2 + 128*128;     __bf16* c3Ws = c3W3 + 128*128;
  prep(F(40), c3W1, 256, 256); prep(F(46), c3W2, 128, 128);
  prep(F(52), c3W3, 128, 128); prep(F(58), c3Ws, 128, 128);

  k_gnproj<<<G, 256, 0, stream>>>(F(0), F(4), F(5), F(6), projT,
                                  F(8), F(10), F(12), F(14), q);

  k_logits<<<E / 8, 256, 0, stream>>>(q, kk, src, dst, logit, mkey);
  k_attn<<<E / 8, 256, 0, stream>>>(logit, mkey, vv, src, dst, den, aggub);

  ConvP c2{c2W1, c2W2, c2W3, c2Ws,
           F(17), F(18), F(19), F(20), F(21),
           F(23), F(24), F(25), F(26), F(27),
           F(29), F(30), F(31), F(32), F(33),
           F(35), F(36), F(37), F(38), F(39)};
  ConvP c3{c3W1, c3W2, c3W3, c3Ws,
           F(41), F(42), F(43), F(44), F(45),
           F(47), F(48), F(49), F(50), F(51),
           F(53), F(54), F(55), F(56), F(57),
           F(59), F(60), F(61), F(62), F(63)};
  HeadP hp{F(64), F(65), F(66), F(67), F(68), F(69), F(70), F(71), F(72), F(73)};

  constexpr unsigned SMEM =
      (4u * TILE + 128u + 512u + 1024u + 512u + 160u + 64u) * sizeof(float);
  (void)hipFuncSetAttribute(reinterpret_cast<const void*>(k_graph),
                            hipFuncAttributeMaxDynamicSharedMemorySize, (int)SMEM);
  k_graph<<<G, 256, SMEM, stream>>>(aggub, den, xr, F(15), F(2), c2, c3, hp,
                                    (float*)d_out);
}
